// JBF_block_9526237462697
// MI455X (gfx1250) — matrix operand order repaired and checked by reading, not compiled, not run on hardware
//
#include <hip/hip_runtime.h>
#include <hip/hip_bf16.h>

// ---------------------------------------------------------------------------
// Learned joint-bilateral filter on gfx1250 (MI455X).
//
// Range branch (the heavy part, 147456 samples) is lowered to two dense
// GEMMs executed with V_WMMA_F32_16X16X32_BF16:
//   Y1(125x16) = W1(125x343) x G(343x16)      [pad 128x352, K-step 32]
//   Y2( 27x16) = W2( 27x125) x Y1             [pad  32x128, K-step 32]
// W1/W2 are Toeplitz embeddings of the 27 conv taps, pre-baked into the
// exact WMMA A-operand fragment layout (bf16) by a setup kernel, so the
// main kernel's A loads are contiguous 32B/lane and L2-resident.
// ---------------------------------------------------------------------------

typedef __attribute__((ext_vector_type(16))) __bf16 v16bf;
typedef __attribute__((ext_vector_type(8)))  __bf16 v8bf;
typedef __attribute__((ext_vector_type(8)))  float  v8f;

#define DD    20
#define HH    96
#define WWID  96
#define DOUT  16
#define NSAMP (DOUT * HH * WWID)        // 147456

#define K1    343                        // 7*7*7
#define M1    125                        // 5*5*5
#define K2    125
#define M2    27
#define KSTEP1 11                        // ceil(343/32) -> 352
#define MT1    8                         // ceil(125/16) -> 128
#define KSTEP2 4                         // 125 -> 128
#define MT2    2                         // 27  -> 32

#define W1F_HALVES (MT1 * KSTEP1 * 32 * 16)   // 45056
#define W2F_HALVES (MT2 * KSTEP2 * 32 * 16)   //  4096

// A-operand (16x32 bf16) element->K mapping (contiguous, matches B):
//   lanes 0-15 : halves 0..15 -> K 0..15
//   lanes 16-31: halves 0..15 -> K 16..31
__device__ __forceinline__ int a_klocal(int h, int lane) {
    return h + (lane >= 16 ? 16 : 0);
}

// ---------------------------------------------------------------------------
// Setup kernel: domain branch (f32) + fragment-ready W1/W2 (bf16).
// ---------------------------------------------------------------------------
__global__ __launch_bounds__(256) void jbf_setup(
    const float* __restrict__ dn,                      // domain_neighbor (343)
    const float* __restrict__ w1d, const float* __restrict__ b1d,
    const float* __restrict__ w2d, const float* __restrict__ b2d,
    const float* __restrict__ w1r, const float* __restrict__ w2r,
    float* __restrict__ ws_dom,                        // 27 floats out
    unsigned short* __restrict__ w1f,                  // W1F_HALVES bf16 out
    unsigned short* __restrict__ w2f)                  // W2F_HALVES bf16 out
{
    __shared__ float s_y1[M1];
    const int tid = threadIdx.x;

    // ---- domain branch, conv1: 7^3 -> 5^3 ----
    if (tid < M1) {
        const int od = tid / 25, oh = (tid / 5) % 5, ow = tid % 5;
        float acc = b1d[0];
        for (int kd = 0; kd < 3; ++kd)
            for (int kh = 0; kh < 3; ++kh)
                for (int kw = 0; kw < 3; ++kw)
                    acc += w1d[kd * 9 + kh * 3 + kw] *
                           dn[(od + kd) * 49 + (oh + kh) * 7 + (ow + kw)];
        s_y1[tid] = fmaxf(acc, 0.0f);
    }
    __syncthreads();
    // ---- domain branch, conv2: 5^3 -> 3^3 ----
    if (tid < M2) {
        const int od = tid / 9, oh = (tid / 3) % 3, ow = tid % 3;
        float acc = b2d[0];
        for (int kd = 0; kd < 3; ++kd)
            for (int kh = 0; kh < 3; ++kh)
                for (int kw = 0; kw < 3; ++kw)
                    acc += w2d[kd * 9 + kh * 3 + kw] *
                           s_y1[(od + kd) * 25 + (oh + kh) * 5 + (ow + kw)];
        ws_dom[tid] = fmaxf(acc, 0.0f);
    }

    // ---- W1 fragments: frag f = t*KSTEP1+s, entry ((f*32)+lane)*16 + h ----
    for (int idx = tid; idx < W1F_HALVES; idx += blockDim.x) {
        const int h    = idx & 15;
        const int lane = (idx >> 4) & 31;
        const int f    = idx >> 9;
        const int t    = f / KSTEP1, s = f % KSTEP1;
        const int m    = t * 16 + (lane & 15);
        const int k    = s * 32 + a_klocal(h, lane);
        float v = 0.0f;
        if (m < M1 && k < K1) {
            const int od = m / 25, oh = (m / 5) % 5, ow = m % 5;
            const int jd = k / 49, jh = (k / 7) % 7, jw = k % 7;
            const int kd = jd - od, kh = jh - oh, kw = jw - ow;
            if ((unsigned)kd < 3u && (unsigned)kh < 3u && (unsigned)kw < 3u)
                v = w1r[kd * 9 + kh * 3 + kw];
        }
        const __bf16 bv = (__bf16)v;
        w1f[idx] = __builtin_bit_cast(unsigned short, bv);
    }

    // ---- W2 fragments ----
    for (int idx = tid; idx < W2F_HALVES; idx += blockDim.x) {
        const int h    = idx & 15;
        const int lane = (idx >> 4) & 31;
        const int f    = idx >> 9;
        const int t    = f / KSTEP2, s = f % KSTEP2;
        const int m    = t * 16 + (lane & 15);
        const int k    = s * 32 + a_klocal(h, lane);
        float v = 0.0f;
        if (m < M2 && k < K2) {
            const int od = m / 9,  oh = (m / 3) % 3, ow = m % 3;
            const int jd = k / 25, jh = (k / 5) % 5, jw = k % 5;
            const int kd = jd - od, kh = jh - oh, kw = jw - ow;
            if ((unsigned)kd < 3u && (unsigned)kh < 3u && (unsigned)kw < 3u)
                v = w2r[kd * 9 + kh * 3 + kw];
        }
        const __bf16 bv = (__bf16)v;
        w2f[idx] = __builtin_bit_cast(unsigned short, bv);
    }
}

// ---------------------------------------------------------------------------
// Main kernel: 8 waves/block, each wave owns 16 samples (one WMMA N-tile).
// ---------------------------------------------------------------------------
__global__ __launch_bounds__(256) void jbf_main(
    const float* __restrict__ x,                       // (20,96,96)
    const float* __restrict__ guide,                   // (N,343)
    const float* __restrict__ b1r, const float* __restrict__ b2r,
    const float* __restrict__ dom,                     // 27 floats
    const unsigned short* __restrict__ w1f,
    const unsigned short* __restrict__ w2f,
    float* __restrict__ out)                           // (N)
{
    // Y1 transposed per wave: [sample 0..15][k 0..127], bf16
    __shared__ alignas(32) unsigned short y1[8][16][128];   // 32 KB
    __shared__ float y2[8][16][32];                         // 16 KB

    const int lane = threadIdx.x & 31;
    const int wv   = threadIdx.x >> 5;
    const int nloc = lane & 15;          // sample / N-column for this lane
    const int hi   = lane >> 4;          // lane-half (K split for B operand)
    const int n    = (blockIdx.x * 8 + wv) * 16 + nloc;

    const float* gbase = guide + (size_t)n * K1;

    // ---------------- GEMM1: Y1 = relu(W1 * G + b1) ----------------
    v8f acc[MT1];
    #pragma unroll
    for (int t = 0; t < MT1; ++t)
        #pragma unroll
        for (int r = 0; r < 8; ++r) acc[t][r] = 0.0f;

    #pragma unroll
    for (int s = 0; s < KSTEP1; ++s) {
        // B operand (32x16 bf16): lanes 0-15 K=0..15, lanes 16-31 K=16..31,
        // contiguous K per lane -> 16 consecutive guide floats.
        v16bf bfr;
        #pragma unroll
        for (int h = 0; h < 16; ++h) {
            const int k = s * 32 + hi * 16 + h;
            bfr[h] = (k < K1) ? (__bf16)gbase[k] : (__bf16)0.0f;
        }
        #pragma unroll
        for (int t = 0; t < MT1; ++t) {
            const v16bf afr = *reinterpret_cast<const v16bf*>(
                w1f + ((size_t)(t * KSTEP1 + s) * 32 + lane) * 16);
            acc[t] = __builtin_amdgcn_wmma_f32_16x16x32_bf16(
                false, afr, false, bfr, (short)0, acc[t], false, false);
        }
    }

    // bias + relu, store Y1^T (D layout: lane holds rows m0..m0+7, col nloc)
    const float bias1 = b1r[0];
    #pragma unroll
    for (int t = 0; t < MT1; ++t) {
        const int m0 = t * 16 + hi * 8;
        v8bf pk;
        #pragma unroll
        for (int r = 0; r < 8; ++r)
            pk[r] = (__bf16)fmaxf(acc[t][r] + bias1, 0.0f);
        *reinterpret_cast<v8bf*>(&y1[wv][nloc][m0]) = pk;
    }
    __syncthreads();

    // ---------------- GEMM2: Y2 = relu(W2 * Y1 + b2) ----------------
    v8f acc2[MT2];
    #pragma unroll
    for (int t = 0; t < MT2; ++t)
        #pragma unroll
        for (int r = 0; r < 8; ++r) acc2[t][r] = 0.0f;

    #pragma unroll
    for (int s = 0; s < KSTEP2; ++s) {
        const v16bf bfr = *reinterpret_cast<const v16bf*>(
            &y1[wv][nloc][s * 32 + hi * 16]);            // 32B contiguous
        #pragma unroll
        for (int t = 0; t < MT2; ++t) {
            const v16bf afr = *reinterpret_cast<const v16bf*>(
                w2f + ((size_t)(t * KSTEP2 + s) * 32 + lane) * 16);
            acc2[t] = __builtin_amdgcn_wmma_f32_16x16x32_bf16(
                false, afr, false, bfr, (short)0, acc2[t], false, false);
        }
    }

    const float bias2 = b2r[0];
    #pragma unroll
    for (int t = 0; t < MT2; ++t) {
        const int m0 = t * 16 + hi * 8;
        #pragma unroll
        for (int r = 0; r < 8; ++r)
            y2[wv][nloc][m0 + r] = fmaxf(acc2[t][r] + bias2, 0.0f);
    }
    __syncthreads();

    // ---------------- normalized 27-tap weighted sum ----------------
    // weights[k] = dom[k]*y2[k] + 1e-10 ; patch value from x with H/W zero
    // pad, depth slices d+1+kd (always in range). Lane pair (nloc,hi) splits
    // the 27 taps 14/13 and reduces via shfl_xor(16).
    const int d  = n / (HH * WWID);
    const int rm = n % (HH * WWID);
    const int h0 = rm / WWID, w0 = rm % WWID;

    float numer = 0.0f, denom = 0.0f;
    const int kbeg = hi ? 14 : 0;
    const int kend = hi ? 27 : 14;
    for (int k = kbeg; k < kend; ++k) {
        const int kd = k / 9, kh = (k / 3) % 3, kw = k % 3;
        const float wgt = dom[k] * y2[wv][nloc][k] + 1e-10f;
        denom += wgt;
        const int dd = d + 1 + kd;
        const int hh = h0 + kh - 1;
        const int ww = w0 + kw - 1;
        float pv = 0.0f;
        if ((unsigned)hh < (unsigned)HH && (unsigned)ww < (unsigned)WWID)
            pv = x[((size_t)dd * HH + hh) * WWID + ww];
        numer += wgt * pv;
    }
    numer += __shfl_xor(numer, 16, 32);
    denom += __shfl_xor(denom, 16, 32);
    if (hi == 0) out[n] = numer / denom;
}

// ---------------------------------------------------------------------------
extern "C" void kernel_launch(void* const* d_in, const int* in_sizes, int n_in,
                              void* d_out, int out_size, void* d_ws, size_t ws_size,
                              hipStream_t stream) {
    const float* x     = (const float*)d_in[0];
    const float* dn    = (const float*)d_in[1];
    const float* guide = (const float*)d_in[2];
    const float* w1d   = (const float*)d_in[3];
    const float* b1d   = (const float*)d_in[4];
    const float* w2d   = (const float*)d_in[5];
    const float* b2d   = (const float*)d_in[6];
    const float* w1r   = (const float*)d_in[7];
    const float* b1r   = (const float*)d_in[8];
    const float* w2r   = (const float*)d_in[9];
    const float* b2r   = (const float*)d_in[10];

    // workspace layout (all 32B aligned)
    float*          ws_dom = (float*)d_ws;                              // 128 B
    unsigned short* w1f    = (unsigned short*)((char*)d_ws + 128);      // 90112 B
    unsigned short* w2f    = (unsigned short*)((char*)d_ws + 128 + W1F_HALVES * 2);

    jbf_setup<<<1, 256, 0, stream>>>(dn, w1d, b1d, w2d, b2d, w1r, w2r,
                                     ws_dom, w1f, w2f);

    jbf_main<<<NSAMP / 128, 256, 0, stream>>>(x, guide, b1r, b2r,
                                              ws_dom, w1f, w2f,
                                              (float*)d_out);
}
